// RNNNet_82033875353799
// MI455X (gfx1250) — matrix operand order repaired and checked by reading, not compiled, not run on hardware
//
#include <hip/hip_runtime.h>
#include <hip/hip_bf16.h>
#include <math.h>

// ---------------------------------------------------------------------------
// 7-layer tanh-RNN (H=64, T=28, V=28) over N sequences, fused in one kernel.
// Each workgroup (128 threads = 4 wave32) owns 16 sequences and runs all
// layers out of LDS using v_wmma_f32_16x16x32_f16 for both the input and the
// recurrent GEMMs (f16 operands, f32 accumulate). tanh uses the native
// gfx1250 V_TANH_F32 transcendental instead of the branchy libm expansion.
// ---------------------------------------------------------------------------

#define TT 28
#define VV 28
#define HH 64
#define LL 7
#define SSTR 72   // padded row stride (halves) for the sequence buffer
#define WSTR 72   // padded row stride (halves) for weight matrices

typedef __attribute__((ext_vector_type(16))) _Float16 v16h;
typedef __attribute__((ext_vector_type(8)))  _Float16 v8h;
typedef __attribute__((ext_vector_type(8)))  float    v8f;

// Native CDNA5 tanh (TRANS32 op). One instruction vs ~40 branchy libm VALU ops.
__device__ __forceinline__ float fast_tanh(float v) {
#if defined(__has_builtin)
#if __has_builtin(__builtin_amdgcn_tanhf)
    return __builtin_amdgcn_tanhf(v);
#else
    float r;
    // v_nop after the trans op: its result needs one intervening op
    // before use (compiler can't see through the asm).
    asm("v_tanh_f32 %0, %1\n\tv_nop" : "=v"(r) : "v"(v));
    return r;
#endif
#else
    float r;
    asm("v_tanh_f32 %0, %1\n\tv_nop" : "=v"(r) : "v"(v));
    return r;
#endif
}

// 16 contiguous halves -> v16h (16B-aligned ds_load_b128 pairs)
__device__ __forceinline__ v16h ld16(const _Float16* p) {
    v8h a = *(const v8h*)(p);
    v8h b = *(const v8h*)(p + 8);
    v16h r;
#pragma unroll
    for (int i = 0; i < 8; ++i) { r[i] = a[i]; r[i + 8] = b[i]; }
    return r;
}

// A-matrix fragment (16x32 f16, M = batch row, K = feature) from a row-major
// [16][SSTR] f16 LDS tile, K-chunk kc in {0,1}.
// k order used here (contiguous, not the ISA table order): lanes 0-15 (M=lane): K = kc*32 + 0..15
//               lanes 16-31 (M=lane-16): K = kc*32 + 16..31
__device__ __forceinline__ v16h loadA(const _Float16* base, int lane, int kc) {
    const int row = lane & 15;
    const int hi  = lane >> 4;
    const _Float16* p = base + row * SSTR + kc * 32 + hi * 16;
    v8h a = *(const v8h*)(p);
    v8h b = *(const v8h*)(p + 8);
    v16h r;
#pragma unroll
    for (int i = 0; i < 8; ++i) { r[i] = a[i]; r[i + 8] = b[i]; }
    return r;
}

// B-matrix fragment (32x16 f16, B[k][n] = W[n][k]) from row-major weight
// LDS [64][WSTR] (row = output feature n, col = input feature k).
// k order used here (contiguous, not the ISA table order): lanes 0-15 (N=lane):     K = kc*32 + 0..15
//               lanes 16-31 (N=lane-16): K = kc*32 + 16..31
__device__ __forceinline__ v16h loadB(const _Float16* w, int lane, int wt, int kc) {
    const int n  = (lane & 15) + wt * 16;
    const int kb = kc * 32 + (lane >> 4) * 16;
    return ld16(w + n * WSTR + kb);
}

#define WMMA_ACC(A, B, C) \
    __builtin_amdgcn_wmma_f32_16x16x32_f16(false, (A), false, (B), (short)0, (C), false, false)

extern __shared__ __align__(16) unsigned char smem_raw[];

__global__ __launch_bounds__(128, 1)
void rnn7_wmma_kernel(const float* __restrict__ x,     // (N,T,V)
                      const float* __restrict__ Wih0,  // (H,V)
                      const float* __restrict__ Wih,   // (L-1,H,H)
                      const float* __restrict__ Whh,   // (L,H,H)
                      const float* __restrict__ bih,   // (L,H)
                      const float* __restrict__ bhh,   // (L,H)
                      const float* __restrict__ Wc,    // (10,H)
                      const float* __restrict__ bc,    // (10,)
                      float* __restrict__ out,         // (N,10)
                      int N) {
    _Float16* s_seq  = (_Float16*)smem_raw;              // [TT][16][SSTR]
    _Float16* s_wih  = s_seq + TT * 16 * SSTR;           // [64][WSTR]
    _Float16* s_whh  = s_wih + 64 * WSTR;                // [64][WSTR]
    float*    s_bias = (float*)(s_whh + 64 * WSTR);      // [64]

    const int tid  = threadIdx.x;
    const int lane = tid & 31;
    const int wt   = tid >> 5;           // wave id 0..3 -> 16-col output tile
    const int n0   = blockIdx.x * 16;    // first sequence of this tile

    // ---- zero sequence buffer (gives K-padding + dead-row padding) ----
    for (int i = tid; i < TT * 16 * SSTR; i += 128) s_seq[i] = (_Float16)0.f;
    __syncthreads();

    // ---- load x tile (16 seqs x T x V), f32 -> f16, zero-padded to K=64 ----
    for (int i = tid; i < 16 * TT * VV; i += 128) {
        const int m = i / (TT * VV);
        const int r = i % (TT * VV);
        const int t = r / VV;
        const int v = r % VV;
        const int n = n0 + m;
        if (n < N)
            s_seq[(t * 16 + m) * SSTR + v] = (_Float16)x[(size_t)n * (TT * VV) + r];
    }
    __syncthreads();

    // ---- layer loop ----
    for (int l = 0; l < LL; ++l) {
        // stage weights for this layer (Wih0 zero-padded K=28->64 for l==0)
        for (int i = tid; i < 64 * 64; i += 128) {
            const int r = i >> 6, c = i & 63;
            float vi;
            if (l == 0) vi = (c < VV) ? Wih0[r * VV + c] : 0.f;
            else        vi = Wih[(size_t)(l - 1) * 4096 + i];
            s_wih[r * WSTR + c] = (_Float16)vi;
            s_whh[r * WSTR + c] = (_Float16)Whh[(size_t)l * 4096 + i];
        }
        for (int i = tid; i < 64; i += 128)
            s_bias[i] = bih[l * 64 + i] + bhh[l * 64 + i];
        __syncthreads();

        // per-wave loop-invariant B fragments (weights live in VGPRs all layer)
        const v16h bi0 = loadB(s_wih, lane, wt, 0);
        const v16h bi1 = loadB(s_wih, lane, wt, 1);
        const v16h bh0 = loadB(s_whh, lane, wt, 0);
        const v16h bh1 = loadB(s_whh, lane, wt, 1);
        const float bb = s_bias[wt * 16 + (lane & 15)];
        const int   nc = wt * 16 + (lane & 15);   // output column
        const int   mb = (lane >> 4) * 8;         // output row base

        for (int t = 0; t < TT; ++t) {
            _Float16* seqt = s_seq + t * 16 * SSTR;

            // input GEMM: in_t (16x64) @ Wih^T tile
            v16h a0 = loadA(seqt, lane, 0);
            v16h a1 = loadA(seqt, lane, 1);
            v8f acc;
#pragma unroll
            for (int i = 0; i < 8; ++i) acc[i] = bb;
            acc = WMMA_ACC(a0, bi0, acc);
            acc = WMMA_ACC(a1, bi1, acc);

            // recurrent GEMM: h_{t-1} (16x64) @ Whh^T tile  (h_{-1} == 0)
            if (t > 0) {
                const _Float16* seqp = s_seq + (t - 1) * 16 * SSTR;
                v16h h0 = loadA(seqp, lane, 0);
                v16h h1 = loadA(seqp, lane, 1);
                acc = WMMA_ACC(h0, bh0, acc);
                acc = WMMA_ACC(h1, bh1, acc);
            }

            __syncthreads();   // everyone done READING seq[t] / seq[t-1]

            // h_t = tanh(acc), overwrite seq[t] in place (input for layer l+1)
#pragma unroll
            for (int i = 0; i < 8; ++i)
                seqt[(mb + i) * SSTR + nc] = (_Float16)fast_tanh(acc[i]);

            __syncthreads();   // h_t visible before next timestep reads it
        }
    }

    // ---- classifier: out = h_last @ Wc^T + bc  (16x10, tiny -> scalar) ----
    const _Float16* h27 = s_seq + 27 * 16 * SSTR;
    for (int o = tid; o < 16 * 10; o += 128) {
        const int m = o / 10;
        const int c = o % 10;
        const int n = n0 + m;
        if (n < N) {
            float s = bc[c];
#pragma unroll
            for (int k = 0; k < HH; ++k)
                s += (float)h27[m * SSTR + k] * Wc[c * HH + k];
            out[(size_t)n * 10 + c] = s;
        }
    }
}

extern "C" void kernel_launch(void* const* d_in, const int* in_sizes, int n_in,
                              void* d_out, int out_size, void* d_ws, size_t ws_size,
                              hipStream_t stream) {
    const float* x    = (const float*)d_in[0];
    const float* Wih0 = (const float*)d_in[1];
    const float* Wih  = (const float*)d_in[2];
    const float* Whh  = (const float*)d_in[3];
    const float* bih  = (const float*)d_in[4];
    const float* bhh  = (const float*)d_in[5];
    const float* Wc   = (const float*)d_in[6];
    const float* bc   = (const float*)d_in[7];
    float* out = (float*)d_out;

    const int N      = in_sizes[0] / (TT * VV);
    const int ntiles = (N + 15) / 16;

    const size_t smem = (size_t)(TT * 16 * SSTR + 2 * 64 * WSTR) * sizeof(short)
                      + 64 * sizeof(float);   // ~83.2 KB

    // Opt in to >64KB dynamic LDS where the runtime requires it (no-op otherwise).
    (void)hipFuncSetAttribute((const void*)rnn7_wmma_kernel,
                              hipFuncAttributeMaxDynamicSharedMemorySize,
                              (int)smem);

    rnn7_wmma_kernel<<<ntiles, 128, smem, stream>>>(x, Wih0, Wih, Whh, bih, bhh,
                                                    Wc, bc, out, N);
}
